// AttentionDecoderTCN_24206435680837
// MI455X (gfx1250) — matrix operand order repaired and checked by reading, not compiled, not run on hardware
//
#include <hip/hip_runtime.h>
#include <hip/hip_bf16.h>

typedef _Float16 half_t;
typedef __attribute__((ext_vector_type(16))) _Float16 v16h;
typedef __attribute__((ext_vector_type(8)))  float    v8f;

#define TDIM  1500
#define TPAD  1536
#define BATCH 32
#define EDIM  512
#define HTD   512
#define HAD   512
#define KATT  32
#define LDEC  200
#define VOUT  1001
#define CKW   3

// ---------------------------------------------------------------------------
// CDNA5 wave32 WMMA 16x16x32 f16 fragment layouts (ISA 7.12.2):
// A: lane lo=lane&15 holds row M=lo; hi=lane>>4: K={0..7,16..23} / {8..15,24..31}
// B: lane lo holds col N=lo; hi: K=0..15 / 16..31 (code's order, A loaded alike; not the ISA order)
// C: vgpr i -> M = hi*8+i, N = lo
// Packed-B layout: tile (ntile,kt) stored as 32 lanes x 16 contiguous halves:
//   dst[((ntile*Kt + kt)*32 + lane)*16 + e]  where  k = kt*32 + hi*16 + e,
//   n = ntile*16 + lo.  One aligned v16h load (2x b128) per lane per tile.
// ---------------------------------------------------------------------------

__device__ __forceinline__ int a_koff(int v, int hi) {
  return (hi << 4) + (v << 1);
}

__device__ __forceinline__ v16h ldB(const half_t* p) {
  return *reinterpret_cast<const v16h*>(p);
}

template <typename AT>
__device__ __forceinline__ v16h ldA(const AT* Ar, int kt, int hi) {
  v16h a;
#pragma unroll
  for (int v = 0; v < 8; ++v) {
    const AT* ap = Ar + (kt << 5) + a_koff(v, hi);
    a[2 * v]     = (half_t)(float)ap[0];
    a[2 * v + 1] = (half_t)(float)ap[1];
  }
  return a;
}

#define WMMA(A_, B_, C_) \
  __builtin_amdgcn_wmma_f32_16x16x32_f16(false, (A_), false, (B_), (short)0, (C_), false, false)

template <typename OT, int ACT, bool NG>
__device__ __forceinline__ void storeC(v8f acc, OT* __restrict__ C,
                                       const float* __restrict__ bias,
                                       int N, int tm, int col, int hi) {
  const bool cok = (!NG) || (col < N);
  const float bs = cok ? bias[col] : 0.f;
#pragma unroll
  for (int i = 0; i < 8; ++i) {
    float v = acc[i] + bs;
    if (ACT == 1) v = fmaxf(v, 0.f);
    if (cok) C[(size_t)(tm + (hi << 3) + i) * N + col] = (OT)v;
  }
}

// ---------------------------------------------------------------------------
// prep: pack f32 row-major (K x N) weights into WMMA-B fragment layout (f16)
// ---------------------------------------------------------------------------
__global__ void k_packB(const float* __restrict__ src, half_t* __restrict__ dst,
                        int K, int N, int Ntiles) {
  const int Kt = K >> 5;
  size_t tot = (size_t)Ntiles * Kt * 512;
  size_t i = (size_t)blockIdx.x * 256 + threadIdx.x;
  if (i >= tot) return;
  int e      = (int)(i & 15);
  int lane   = (int)((i >> 4) & 31);
  size_t til = i >> 9;
  int kt     = (int)(til % Kt);
  int ntile  = (int)(til / Kt);
  int k = (kt << 5) + ((lane >> 4) << 4) + e;
  int n = (ntile << 4) + (lane & 15);
  dst[i] = (n < N) ? (half_t)src[(size_t)k * N + n] : (half_t)0.f;
}

// tcn_w (6, O, I, CK) -> per (ly,tap) packed B with K=I, N=O (Kt=16, Ntiles=32)
__global__ void k_packTcn(const float* __restrict__ w, half_t* __restrict__ dst) {
  const size_t tot = (size_t)18 * 512 * 512;
  size_t i = (size_t)blockIdx.x * 256 + threadIdx.x;
  if (i >= tot) return;
  int e      = (int)(i & 15);
  int lane   = (int)((i >> 4) & 31);
  size_t til = i >> 9;
  int kt    = (int)(til % 16);
  int ntile = (int)((til / 16) % 32);
  int tap   = (int)(til / 512);
  int k3 = tap % 3, ly = tap / 3;
  int k = (kt << 5) + ((lane >> 4) << 4) + e;
  int n = (ntile << 4) + (lane & 15);
  dst[i] = (half_t)w[(((size_t)ly * HTD + n) * HTD + k) * CKW + k3];
}

__global__ void k_embed(const int* __restrict__ tg, const float* __restrict__ emb,
                        half_t* __restrict__ o) {
  int r = blockIdx.x;
  int t = tg[r];
  for (int c = threadIdx.x; c < HTD; c += 256)
    o[(size_t)r * HTD + c] = (half_t)emb[(size_t)t * HTD + c];
}

__global__ void k_resrelu(const half_t* __restrict__ a, half_t* __restrict__ io) {
  size_t i = (size_t)blockIdx.x * 256 + threadIdx.x;
  io[i] = (half_t)fmaxf((float)a[i] + (float)io[i], 0.f);
}

__global__ void k_comb(const half_t* __restrict__ tcn, const float* __restrict__ ctx,
                       half_t* __restrict__ comb) {
  int r = blockIdx.x;
  for (int c = threadIdx.x; c < HTD + EDIM; c += 256) {
    half_t v = (c < HTD) ? tcn[(size_t)r * HTD + c]
                         : (half_t)ctx[(size_t)r * EDIM + (c - HTD)];
    comb[(size_t)r * (HTD + EDIM) + c] = v;
  }
}

// ---------------------------------------------------------------------------
// WMMA GEMM, 16x64 tile per wave, software-pipelined k-loop (double-buffered
// fragments so loads for kt+1 overlap the four WMMAs of kt).
// ---------------------------------------------------------------------------
template <typename AT, typename OT, int ACT, bool NG>
__global__ void k_gemm(const AT* __restrict__ A, const half_t* __restrict__ Bp,
                       const float* __restrict__ bias, OT* __restrict__ C,
                       int M, int N, int Kt) {
  const int lane = threadIdx.x & 31;
  const int wave = threadIdx.x >> 5;
  const int lo = lane & 15, hi = lane >> 4;
  const int tm = blockIdx.x << 4;
  const int tn = ((blockIdx.y << 2) + wave) << 6;
  const int Kd = Kt << 5;

  v8f acc0 = {}, acc1 = {}, acc2 = {}, acc3 = {};
  const AT* Ar = A + ((size_t)tm + lo) * Kd;
  const half_t* Bw = Bp + ((size_t)(tn >> 4) * Kt << 9) + (lane << 4);
  const size_t bts = (size_t)Kt << 9;

  v16h a = ldA(Ar, 0, hi);
  v16h b0 = ldB(Bw), b1 = ldB(Bw + bts), b2 = ldB(Bw + 2 * bts), b3 = ldB(Bw + 3 * bts);

  for (int kt = 0; kt + 1 < Kt; ++kt) {
    const half_t* bb = Bw + ((size_t)(kt + 1) << 9);
    v16h an = ldA(Ar, kt + 1, hi);
    v16h c0 = ldB(bb), c1 = ldB(bb + bts), c2 = ldB(bb + 2 * bts), c3 = ldB(bb + 3 * bts);
    acc0 = WMMA(a, b0, acc0);
    acc1 = WMMA(a, b1, acc1);
    acc2 = WMMA(a, b2, acc2);
    acc3 = WMMA(a, b3, acc3);
    a = an; b0 = c0; b1 = c1; b2 = c2; b3 = c3;
  }
  acc0 = WMMA(a, b0, acc0);
  acc1 = WMMA(a, b1, acc1);
  acc2 = WMMA(a, b2, acc2);
  acc3 = WMMA(a, b3, acc3);

  storeC<OT, ACT, NG>(acc0, C, bias, N, tm, tn + lo, hi);
  storeC<OT, ACT, NG>(acc1, C, bias, N, tm, tn + 16 + lo, hi);
  storeC<OT, ACT, NG>(acc2, C, bias, N, tm, tn + 32 + lo, hi);
  storeC<OT, ACT, NG>(acc3, C, bias, N, tm, tn + 48 + lo, hi);
}

// ---------------------------------------------------------------------------
// TCN dilated causal conv: 3 shifted banded GEMMs, 16x64 per wave, pipelined.
// ---------------------------------------------------------------------------
__global__ void k_tcn_conv(const half_t* __restrict__ H, const half_t* __restrict__ Wl,
                           const float* __restrict__ bias, half_t* __restrict__ O,
                           int dil) {
  const int lane = threadIdx.x & 31;
  const int wave = threadIdx.x >> 5;
  const int lo = lane & 15, hi = lane >> 4;
  const int tm = blockIdx.x << 4;
  const int tn = ((blockIdx.y << 2) + wave) << 6;
  const int l = tm / BATCH;
  const int Kt = HTD >> 5;
  const size_t bts = (size_t)Kt << 9;

  v8f acc0 = {}, acc1 = {}, acc2 = {}, acc3 = {};
  for (int k3 = 0; k3 < CKW; ++k3) {
    const int sh = (CKW - 1 - k3) * dil;
    if (l < sh) continue;                             // wave-uniform
    const half_t* Ar = H + (((size_t)tm - (size_t)sh * BATCH) + lo) * HTD;
    const half_t* Bw = Wl + (size_t)k3 * (512 * 512)
                         + ((size_t)(tn >> 4) * Kt << 9) + (lane << 4);
    v16h a = ldA(Ar, 0, hi);
    v16h b0 = ldB(Bw), b1 = ldB(Bw + bts), b2 = ldB(Bw + 2 * bts), b3 = ldB(Bw + 3 * bts);
    for (int kt = 0; kt + 1 < Kt; ++kt) {
      const half_t* bb = Bw + ((size_t)(kt + 1) << 9);
      v16h an = ldA(Ar, kt + 1, hi);
      v16h c0 = ldB(bb), c1 = ldB(bb + bts), c2 = ldB(bb + 2 * bts), c3 = ldB(bb + 3 * bts);
      acc0 = WMMA(a, b0, acc0);
      acc1 = WMMA(a, b1, acc1);
      acc2 = WMMA(a, b2, acc2);
      acc3 = WMMA(a, b3, acc3);
      a = an; b0 = c0; b1 = c1; b2 = c2; b3 = c3;
    }
    acc0 = WMMA(a, b0, acc0);
    acc1 = WMMA(a, b1, acc1);
    acc2 = WMMA(a, b2, acc2);
    acc3 = WMMA(a, b3, acc3);
  }

#pragma unroll
  for (int nt = 0; nt < 4; ++nt) {
    const int col = tn + (nt << 4) + lo;
    const float bs = bias[col];
    const v8f acc = (nt == 0) ? acc0 : (nt == 1) ? acc1 : (nt == 2) ? acc2 : acc3;
#pragma unroll
    for (int i = 0; i < 8; ++i) {
      const int row = tm + (hi << 3) + i;
      O[(size_t)row * HTD + col] = (half_t)fmaxf(acc[i] + bs, 0.f);
    }
  }
}

// ---------------------------------------------------------------------------
// init attention scores: s0[t,b] = encoded[t,b,:] . ai_w + ai_b
// ---------------------------------------------------------------------------
__global__ void k_init_score(const float* __restrict__ enc, const float* __restrict__ ai_w,
                             const float* __restrict__ ai_b, float* __restrict__ s0) {
  const int wave = threadIdx.x >> 5, lane = threadIdx.x & 31;
  const int idx = blockIdx.x * 8 + wave;
  if (idx >= TDIM * BATCH) return;
  const float* p = enc + (size_t)idx * EDIM;
  float acc = 0.f;
  for (int j = lane; j < EDIM; j += 32) acc += p[j] * ai_w[j];
#pragma unroll
  for (int o = 16; o; o >>= 1) acc += __shfl_xor(acc, o, 32);
  if (lane == 0) s0[idx] = acc + ai_b[0];
}

// ---------------------------------------------------------------------------
// masked softmax over T (per b); outputs transposed f32 (B,T) and f16 (B,TPAD)
// ---------------------------------------------------------------------------
__global__ void k_softmax(const float* __restrict__ s, const int* __restrict__ lens,
                          float* __restrict__ att_fT, half_t* __restrict__ att_hT) {
  __shared__ float red[256];
  const int b = blockIdx.x, tid = threadIdx.x;
  const int len = lens[b];
  float mx = -3.0e38f;
  for (int t = tid; t < TDIM; t += 256) {
    float v = s[(size_t)t * BATCH + b] + (t < len ? 0.f : -1e5f);
    mx = fmaxf(mx, v);
  }
  red[tid] = mx; __syncthreads();
  for (int o = 128; o; o >>= 1) { if (tid < o) red[tid] = fmaxf(red[tid], red[tid + o]); __syncthreads(); }
  mx = red[0]; __syncthreads();
  float sm = 0.f;
  for (int t = tid; t < TDIM; t += 256) {
    float v = s[(size_t)t * BATCH + b] + (t < len ? 0.f : -1e5f);
    sm += __expf(v - mx);
  }
  red[tid] = sm; __syncthreads();
  for (int o = 128; o; o >>= 1) { if (tid < o) red[tid] += red[tid + o]; __syncthreads(); }
  const float inv = 1.f / red[0];
  for (int t = tid; t < TDIM; t += 256) {
    float v = s[(size_t)t * BATCH + b] + (t < len ? 0.f : -1e5f);
    float a = __expf(v - mx) * inv;
    att_fT[(size_t)b * TDIM + t] = a;
    att_hT[(size_t)b * TPAD + t] = (half_t)a;
  }
}

// ---------------------------------------------------------------------------
// fused score: one wave per (16-t tile, b). loc-conv = ONE wmma per ha-tile
// (K == KATT == 32); kern fragments pipelined; tanh + as_w dot on the
// accumulator; 16-lane reduce -> s[t,b].
// ---------------------------------------------------------------------------
__global__ void k_score(const half_t* __restrict__ ench, const half_t* __restrict__ att_hT,
                        const half_t* __restrict__ kern, const float* __restrict__ g,
                        const float* __restrict__ as_w, const float* __restrict__ as_b,
                        float* __restrict__ s) {
  const int lane = threadIdx.x & 31;
  const int lo = lane & 15, hi = lane >> 4;
  const int tt = blockIdx.x << 4;
  const int b  = blockIdx.y;

  v16h a;
  const half_t* attb = att_hT + (size_t)b * TPAD;
  const int tb = tt + lo - (KATT - 1) + (hi << 4);
  if (tt >= KATT) {
    const half_t* p0 = attb + tb;
#pragma unroll
    for (int e = 0; e < 8; ++e) { a[e] = p0[e]; a[8 + e] = p0[8 + e]; }
  } else {
#pragma unroll
    for (int e = 0; e < 8; ++e) {
      int t0 = tb + e, t1 = tb + 8 + e;
      a[e]     = (t0 >= 0) ? attb[t0] : (half_t)0.f;
      a[8 + e] = (t1 >= 0) ? attb[t1] : (half_t)0.f;
    }
  }

  float acc[8] = {0.f, 0.f, 0.f, 0.f, 0.f, 0.f, 0.f, 0.f};
  const half_t* kbase = kern + (size_t)b * (KATT * HAD) + ((size_t)lo << 5) + (hi << 4);
  v16h bf = ldB(kbase);
  for (int nt = 0; nt < HAD / 16; ++nt) {
    v16h bn;
    if (nt + 1 < HAD / 16) bn = ldB(kbase + ((size_t)(nt + 1) << 9));   // next ha-tile
    v8f c = {};
    c = WMMA(a, bf, c);
    const int ha = (nt << 4) + lo;
    const float gv = g[(size_t)b * HAD + ha];
    const float aw = as_w[ha];
#pragma unroll
    for (int i = 0; i < 8; ++i) {
      const int t = tt + (hi << 3) + i;
      if (t < TDIM) {
        const float e = (float)ench[((size_t)t * BATCH + b) * HAD + ha];
        acc[i] += aw * tanhf(e + c[i] + gv);
      }
    }
    bf = bn;
  }
#pragma unroll
  for (int off = 1; off < 16; off <<= 1)
#pragma unroll
    for (int i = 0; i < 8; ++i) acc[i] += __shfl_xor(acc[i], off, 32);
  if (lo == 0) {
    const float sb = as_b[0];
#pragma unroll
    for (int i = 0; i < 8; ++i) {
      const int t = tt + (hi << 3) + i;
      if (t < TDIM) s[(size_t)t * BATCH + b] = acc[i] + sb;
    }
  }
}

// ---------------------------------------------------------------------------
// context: ctx[b,e] = sum_t att[b,t]*encoded[t,b,e]
// ---------------------------------------------------------------------------
__global__ void k_ctx(const float* __restrict__ attT, const float* __restrict__ enc,
                      float* __restrict__ ctx) {
  const int b = blockIdx.x, tid = threadIdx.x;
  const int e0 = tid * 2;
  const float* ab = attT + (size_t)b * TDIM;
  float a0 = 0.f, a1 = 0.f;
  for (int t = 0; t < TDIM; ++t) {
    const float a = ab[t];
    const float* p = enc + ((size_t)t * BATCH + b) * EDIM + e0;
    a0 += a * p[0]; a1 += a * p[1];
  }
  ctx[(size_t)b * EDIM + e0]     = a0;
  ctx[(size_t)b * EDIM + e0 + 1] = a1;
}

// ---------------------------------------------------------------------------
// host
// ---------------------------------------------------------------------------
static inline size_t alignup(size_t x) { return (x + 255) & ~(size_t)255; }

extern "C" void kernel_launch(void* const* d_in, const int* in_sizes, int n_in,
                              void* d_out, int out_size, void* d_ws, size_t ws_size,
                              hipStream_t stream) {
  const float* encoded = (const float*)d_in[0];
  const int*   lens    = (const int*)d_in[1];
  const int*   targets = (const int*)d_in[2];
  const float* emb   = (const float*)d_in[3];
  const float* ae_w  = (const float*)d_in[4];  const float* ae_b  = (const float*)d_in[5];
  const float* ai_w  = (const float*)d_in[6];  const float* ai_b  = (const float*)d_in[7];
  const float* ak_w  = (const float*)d_in[8];  const float* ak_b  = (const float*)d_in[9];
  const float* ag_w  = (const float*)d_in[10]; const float* ag_b  = (const float*)d_in[11];
  const float* as_w  = (const float*)d_in[12]; const float* as_b  = (const float*)d_in[13];
  const float* tcn_w = (const float*)d_in[14]; const float* tcn_b = (const float*)d_in[15];
  const float* c1_w  = (const float*)d_in[16]; const float* c1_b  = (const float*)d_in[17];
  const float* c2_w  = (const float*)d_in[18]; const float* c2_b  = (const float*)d_in[19];
  const float* out_w = (const float*)d_in[20]; const float* out_b = (const float*)d_in[21];
  float* out = (float*)d_out;
  (void)in_sizes; (void)n_in; (void)out_size; (void)ws_size;

  char* base = (char*)d_ws; size_t off = 0;
  auto alloc = [&](size_t bytes) -> void* { void* r = base + off; off = alignup(off + bytes); return r; };

  half_t* ench  = (half_t*)alloc((size_t)TDIM * BATCH * HAD * 2);
  half_t* ae_p  = (half_t*)alloc((size_t)32 * 16 * 512 * 2);
  half_t* ak_p  = (half_t*)alloc((size_t)1024 * 16 * 512 * 2);
  half_t* ag_p  = (half_t*)alloc((size_t)32 * 16 * 512 * 2);
  half_t* c1_p  = (half_t*)alloc((size_t)16 * 32 * 512 * 2);
  half_t* c2_p  = (half_t*)alloc((size_t)16 * 8 * 512 * 2);
  half_t* ow_p  = (half_t*)alloc((size_t)64 * 8 * 512 * 2);
  half_t* wtc   = (half_t*)alloc((size_t)18 * 512 * 512 * 2);
  half_t* hbuf  = (half_t*)alloc((size_t)LDEC * BATCH * HTD * 2);
  half_t* tmp1  = (half_t*)alloc((size_t)LDEC * BATCH * HTD * 2);
  half_t* tmp2  = (half_t*)alloc((size_t)LDEC * BATCH * HTD * 2);
  half_t* att_h = (half_t*)alloc((size_t)BATCH * TPAD * 2);
  half_t* kern  = (half_t*)alloc((size_t)BATCH * KATT * HAD * 2);
  half_t* comb  = (half_t*)alloc((size_t)LDEC * BATCH * (HTD + EDIM) * 2);
  half_t* m1    = (half_t*)alloc((size_t)LDEC * BATCH * 256 * 2);
  half_t* m2    = (half_t*)alloc((size_t)LDEC * BATCH * 256 * 2);
  float*  sbuf  = (float*)alloc((size_t)TDIM * BATCH * 4);
  float*  att_f = (float*)alloc((size_t)BATCH * TDIM * 4);
  float*  gbuf  = (float*)alloc((size_t)BATCH * HAD * 4);
  float*  ctxs  = (float*)alloc((size_t)LDEC * BATCH * EDIM * 4);

  auto packB = [&](const float* s, half_t* d, int K, int N, int Ntiles) {
    size_t n = (size_t)Ntiles * (K >> 5) * 512;
    k_packB<<<dim3((unsigned)((n + 255) / 256)), 256, 0, stream>>>(s, d, K, N, Ntiles);
  };
  packB(ae_w, ae_p, 512, 512, 32);
  packB(ak_w, ak_p, 512, 16384, 1024);
  packB(ag_w, ag_p, 512, 512, 32);
  packB(c1_w, c1_p, 1024, 256, 16);
  packB(c2_w, c2_p, 256, 256, 16);
  packB(out_w, ow_p, 256, VOUT, 64);
  {
    size_t n = (size_t)18 * 512 * 512;
    k_packTcn<<<dim3((unsigned)((n + 255) / 256)), 256, 0, stream>>>(tcn_w, wtc);
  }

  // ---- embed + TCN ----
  k_embed<<<dim3(LDEC * BATCH), 256, 0, stream>>>(targets, emb, hbuf);
  const int dils[3] = {1, 2, 4};
  const int MROW = LDEC * BATCH;
  for (int blk = 0; blk < 3; ++blk) {
    const int d = dils[blk];
    k_tcn_conv<<<dim3(MROW / 16, 2), 128, 0, stream>>>(
        hbuf, wtc + (size_t)(2 * blk) * 3 * 512 * 512, tcn_b + (2 * blk) * HTD, tmp1, d);
    k_tcn_conv<<<dim3(MROW / 16, 2), 128, 0, stream>>>(
        tmp1, wtc + (size_t)(2 * blk + 1) * 3 * 512 * 512, tcn_b + (2 * blk + 1) * HTD, tmp2, d);
    k_resrelu<<<dim3((MROW * HTD) / 256), 256, 0, stream>>>(tmp2, hbuf);
  }

  // ---- enc_contrib = encoded @ ae_w + ae_b -> f16 (L2-resident) ----
  k_gemm<float, half_t, 0, false><<<dim3((TDIM * BATCH) / 16, 2), 128, 0, stream>>>(
      encoded, ae_p, ae_b, ench, TDIM * BATCH, HAD, EDIM / 32);

  // ---- init attention ----
  k_init_score<<<dim3((TDIM * BATCH) / 8), 256, 0, stream>>>(encoded, ai_w, ai_b, sbuf);
  k_softmax<<<dim3(BATCH), 256, 0, stream>>>(sbuf, lens, att_f, att_h);

  // ---- sequential location-attention scan ----
  const int TT = (TDIM + 15) / 16;
  for (int l = 0; l < LDEC; ++l) {
    const half_t* lm = hbuf + (size_t)l * BATCH * HTD;
    k_gemm<half_t, half_t, 0, false><<<dim3(BATCH / 16, 64), 128, 0, stream>>>(
        lm, ak_p, ak_b, kern, BATCH, KATT * HAD, HTD / 32);
    k_gemm<half_t, float, 0, false><<<dim3(BATCH / 16, 2), 128, 0, stream>>>(
        lm, ag_p, ag_b, gbuf, BATCH, HAD, HTD / 32);
    k_score<<<dim3(TT, BATCH), 32, 0, stream>>>(ench, att_h, kern, gbuf, as_w, as_b, sbuf);
    k_softmax<<<dim3(BATCH), 256, 0, stream>>>(sbuf, lens, att_f, att_h);
    k_ctx<<<dim3(BATCH), 256, 0, stream>>>(att_f, encoded, ctxs + (size_t)l * BATCH * EDIM);
  }

  // ---- concat + output MLP ----
  k_comb<<<dim3(MROW), 256, 0, stream>>>(hbuf, ctxs, comb);
  k_gemm<half_t, half_t, 1, false><<<dim3(MROW / 16, 1), 128, 0, stream>>>(
      comb, c1_p, c1_b, m1, MROW, 256, (HTD + EDIM) / 32);
  k_gemm<half_t, half_t, 1, false><<<dim3(MROW / 16, 1), 128, 0, stream>>>(
      m1, c2_p, c2_b, m2, MROW, 256, 256 / 32);
  k_gemm<half_t, float, 0, true><<<dim3(MROW / 16, 4), 128, 0, stream>>>(
      m2, ow_p, out_b, out, MROW, VOUT, 256 / 32);
}
